// get_attention_11098195493366
// MI455X (gfx1250) — matrix operand order repaired and checked by reading, not compiled, not run on hardware
//
#include <hip/hip_runtime.h>
#include <cstdint>

// Problem constants (match reference)
#define B_ 4
#define L_ 512
#define D_ 128
#define C_ 128
#define EPS_ 1e-7f
#define NROWS_ (B_ * L_)

typedef __attribute__((ext_vector_type(16))) __bf16 v16bf;
typedef __attribute__((ext_vector_type(8)))  float v8f;
typedef unsigned int v4u __attribute__((ext_vector_type(4)));
typedef int          v8i __attribute__((ext_vector_type(8)));
typedef int          v4i __attribute__((ext_vector_type(4)));

// Tensor Data Mover availability (device pass only; host pass sees 0)
#if defined(__gfx1250__) && __has_builtin(__builtin_amdgcn_tensor_load_to_lds) && \
    __has_builtin(__builtin_amdgcn_s_wait_tensorcnt)
#define HAS_TDM 1
#else
#define HAS_TDM 0
#endif

// ---------- helpers ----------

// float -> bf16 with round-to-nearest-even
__device__ inline uint16_t f2bf(float f) {
    uint32_t u = __builtin_bit_cast(uint32_t, f);
    u += 0x7FFFu + ((u >> 16) & 1u);
    return (uint16_t)(u >> 16);
}

__device__ inline v16bf pack_bf(const uint32_t* w) {
    union { uint32_t u[8]; v16bf v; } t;
#pragma unroll
    for (int i = 0; i < 8; ++i) t.u[i] = w[i];
    return t.v;
}

#define TWO_LOG2E 2.885390081777927f
#define LOG2E     1.442695040888963f

// gfx1250 native V_TANH_F32 when available; exp2+rcp fallback otherwise.
#if __has_builtin(__builtin_amdgcn_tanhf)
__device__ inline float fast_tanh(float x) { return __builtin_amdgcn_tanhf(x); }
#else
__device__ inline float fast_tanh(float x) {
    float e = __builtin_amdgcn_exp2f(x * TWO_LOG2E);
    return 1.0f - 2.0f * __builtin_amdgcn_rcpf(e + 1.0f);
}
#endif
__device__ inline float fast_exp(float x) {
    return __builtin_amdgcn_exp2f(x * LOG2E);
}

// ---------- kernel 1: qb = X@Wt + bxh ; kx = X@Wx  (bf16 WMMA) ----------
// grid = (B*L)/16 blocks, 256 threads (8 waves); wave w owns N-tile w (16 cols).
__global__ __launch_bounds__(256) void proj_kernel(
    const float* __restrict__ inps, const float* __restrict__ wei_t,
    const float* __restrict__ wei_x, const float* __restrict__ bxh,
    float* __restrict__ qb, float* __restrict__ kxo) {

    __shared__ uint16_t sW[D_ * C_];    // 32KB: one weight matrix in bf16
    __shared__ uint16_t sA[16 * D_];    // 4KB : 16-row input tile in bf16

    const int tid = threadIdx.x, lane = tid & 31, wave = tid >> 5;
    const int half = lane >> 4, M = lane & 15;
    const int rowTile = blockIdx.x;                 // over flattened B*L rows
    const float* Arows = inps + (size_t)rowTile * 16 * D_;

    for (int e = tid; e < 16 * D_; e += 256) sA[e] = f2bf(Arows[e]);
    for (int e = tid; e < D_ * C_; e += 256) sW[e] = f2bf(wei_t[e]);
    __syncthreads();

    // A fragments for 4 K-steps of 32, kept in registers (shared by both GEMMs).
    uint32_t afrag[4][8];
    const uint32_t* sA32 = (const uint32_t*)sA;
#pragma unroll
    for (int kk = 0; kk < 4; ++kk)
#pragma unroll
        for (int r = 0; r < 8; ++r) {
            int k = kk * 32 + (r < 4 ? half * 8 + 2 * r : 16 + half * 8 + 2 * (r - 4));
            afrag[kk][r] = sA32[(M * D_ + k) >> 1];
        }

    const int ncol = wave * 16 + M;

    // ---- qt (+bxh fold) ----
    {
        v8f acc = {};
#pragma unroll
        for (int kk = 0; kk < 4; ++kk) {
            uint32_t bfrag[8];
#pragma unroll
            for (int r = 0; r < 8; ++r) {   // B 32x16: lane half picks K 8h..8h+7 and 16+8h..16+8h+7
                int k = kk * 32 + (r < 4 ? half * 8 + 2 * r : 16 + half * 8 + 2 * (r - 4));
                uint32_t lo = sW[k * C_ + ncol];
                uint32_t hi = sW[(k + 1) * C_ + ncol];
                bfrag[r] = lo | (hi << 16);
            }
            acc = __builtin_amdgcn_wmma_f32_16x16x32_bf16(
                false, pack_bf(afrag[kk]), false, pack_bf(bfrag),
                (short)0, acc, false, false);
        }
        float bias = bxh[ncol];
#pragma unroll
        for (int r = 0; r < 8; ++r) {
            int m = r + 8 * half;   // C/D layout: VGPR r -> M = r (+8 for hi half)
            qb[((size_t)rowTile * 16 + m) * C_ + ncol] = acc[r] + bias;
        }
    }

    __syncthreads();
    for (int e = tid; e < D_ * C_; e += 256) sW[e] = f2bf(wei_x[e]);
    __syncthreads();

    // ---- kx ----
    {
        v8f acc = {};
#pragma unroll
        for (int kk = 0; kk < 4; ++kk) {
            uint32_t bfrag[8];
#pragma unroll
            for (int r = 0; r < 8; ++r) {
                int k = kk * 32 + (r < 4 ? half * 8 + 2 * r : 16 + half * 8 + 2 * (r - 4));
                uint32_t lo = sW[k * C_ + ncol];
                uint32_t hi = sW[(k + 1) * C_ + ncol];
                bfrag[r] = lo | (hi << 16);
            }
            acc = __builtin_amdgcn_wmma_f32_16x16x32_bf16(
                false, pack_bf(afrag[kk]), false, pack_bf(bfrag),
                (short)0, acc, false, false);
        }
#pragma unroll
        for (int r = 0; r < 8; ++r) {
            int m = r + 8 * half;
            kxo[((size_t)rowTile * 16 + m) * C_ + ncol] = acc[r];
        }
    }
}

// ---------- kernel 2: fused tanh-scores + online softmax + P@V (WMMA) ----------
// grid = B*(L/16)*nsplit blocks, 256 threads. Block owns (b, 16-query tile, j-split).
// kx chunks are DMA'd into LDS by the Tensor Data Mover (double-buffered) when
// available; the TDM pad feature produces the bank-conflict-free 132-float stride.
#define JC 32
#define KSTRIDE (C_ + 4)   // 132 floats/row (TDM: pad 4 DWORDs every 128 DWORDs)

#if HAS_TDM
// Issue TENSOR_LOAD_TO_LDS for one 32x128 f32 chunk. D# per ISA ch.8:
// 2D tensor == the chunk; data_size=4B; pad_interval=128 DW (code 6),
// pad_amount=4 DW (code 3). Issued by one wave; tracked with TENSORcnt.
__device__ inline void tdm_load_k(const float* gsrc, uint32_t lds_off) {
    uint64_t ga = (uint64_t)(uintptr_t)gsrc;
    v4u g0;
    g0[0] = 1u;                                  // count=1 (valid), user mode
    g0[1] = lds_off;                             // LDS dest byte address
    g0[2] = (uint32_t)ga;                        // global_addr[31:0]
    g0[3] = (uint32_t)(ga >> 32) | (2u << 30);   // global_addr[56:32] | type=2
    v8i g1;
    g1[0] = (int)((2u << 16) |                   // data_size = 4B
                  (1u << 20) |                   // pad_enable
                  (6u << 22) |                   // pad_interval: 128 DWORDs
                  (3u << 25));                   // pad_amount: 4 DWORDs
    g1[1] = (int)(128u << 16);                   // tensor_dim0 = 128 (bits 79:48 lo)
    g1[2] = (int)(32u << 16);                    // tensor_dim1 = 32  (bits 111:80 lo)
    g1[3] = (int)(128u << 16);                   // tile_dim0 = 128   (bits 127:112)
    g1[4] = (int)32;                             // tile_dim1 = 32, tile_dim2 = 0
    g1[5] = (int)128;                            // tensor_dim0_stride = 128
    g1[6] = 0;
    g1[7] = 0;
    v4i z4 = {0, 0, 0, 0};
#if defined(__clang_major__) && (__clang_major__ >= 23)
    v8i z8 = {0, 0, 0, 0, 0, 0, 0, 0};           // amdgpu-toolchain: 6-arg form
    __builtin_amdgcn_tensor_load_to_lds(g0, g1, z4, z4, z8, 0);
#else
    __builtin_amdgcn_tensor_load_to_lds(g0, g1, z4, z4, 0);  // ROCm 7.2: 5-arg
#endif
}
#endif

__global__ __launch_bounds__(256) void attn_kernel(
    const float* __restrict__ inps, const float* __restrict__ qb,
    const float* __restrict__ kx, const float* __restrict__ wei_a,
    float* __restrict__ out, float* __restrict__ pacc,
    float* __restrict__ pm, float* __restrict__ pl, int nsplit) {

#if HAS_TDM
    __shared__ __align__(16) float sK[2][JC * KSTRIDE];   // 33KB double buffer
#else
    __shared__ __align__(16) float sK[1][JC * KSTRIDE];   // 16.5KB
#endif
    __shared__ float    sQ[16 * C_];          // 8KB  qb tile (broadcast reads)
    __shared__ float    sWa[C_];              // 512B
    __shared__ uint16_t sV[JC * D_];          // 8KB  inps chunk, bf16 (WMMA B)
    __shared__ float    sS[16 * JC];          // 2KB  raw scores
    __shared__ uint16_t sP[16 * JC];          // 1KB  probs bf16 (WMMA A)
    __shared__ float    sScale[16], sM[16], sLs[16];

    const int tid = threadIdx.x, lane = tid & 31, wave = tid >> 5;
    const int half = lane >> 4, M = lane & 15;
    const int sp = blockIdx.x % nsplit;
    const int bi = blockIdx.x / nsplit;
    const int b  = bi / (L_ / 16);
    const int it = bi % (L_ / 16);
    const int nch = (L_ / JC) / nsplit;       // chunks handled by this block
    const int jc0 = sp * nch;

    const float* Q = qb + ((size_t)b * L_ + it * 16) * C_;
    for (int e = tid; e < 16 * C_; e += 256) sQ[e] = Q[e];
    for (int e = tid; e < C_; e += 256) sWa[e] = wei_a[e];
    if (tid < 16) { sM[tid] = -3.0e38f; sLs[tid] = 0.0f; }

    v8f acc = {};   // 16x16 f32 accumulator, wave's N-tile of the output
    const float* kbase = kx + (size_t)b * L_ * C_;

#if HAS_TDM
    // prime the pipeline: DMA first chunk while we fall into the loop
    if (wave == 0)
        tdm_load_k(kbase + (size_t)jc0 * JC * C_, (uint32_t)(uintptr_t)&sK[0][0]);
#endif
    __syncthreads();

    for (int jc = jc0; jc < jc0 + nch; ++jc) {
        const float* Kc = kbase + (size_t)jc * JC * C_;
        const float* Vc = inps + ((size_t)b * L_ + jc * JC) * D_;
        int cur;
#if HAS_TDM
        cur = (jc - jc0) & 1;
        if (wave == 0) __builtin_amdgcn_s_wait_tensorcnt(0);  // chunk `cur` landed
        __syncthreads();            // publish TDM writes; prior readers of buf^1 done
        if (wave == 0 && jc + 1 < jc0 + nch)                  // overlap next DMA
            tdm_load_k(Kc + JC * C_, (uint32_t)(uintptr_t)&sK[cur ^ 1][0]);
        if (jc + 1 < jc0 + nch)
            __builtin_prefetch((const char*)(Vc + JC * D_) + tid * 64, 0, 1);
        // stage V only (f32 -> bf16 conversion for the WMMA B operand)
        for (int e = tid; e < JC * D_ / 4; e += 256) {
            float4 v4 = ((const float4*)Vc)[e];
            ((uint32_t*)sV)[2 * e]     = (uint32_t)f2bf(v4.x) | ((uint32_t)f2bf(v4.y) << 16);
            ((uint32_t*)sV)[2 * e + 1] = (uint32_t)f2bf(v4.z) | ((uint32_t)f2bf(v4.w) << 16);
        }
#else
        cur = 0;
        if (jc + 1 < jc0 + nch) {
            __builtin_prefetch((const char*)(Kc + JC * C_) + tid * 64, 0, 1);
            __builtin_prefetch((const char*)(Vc + JC * D_) + tid * 64, 0, 1);
        }
        for (int e = tid; e < JC * C_ / 4; e += 256) {
            int row = e / (C_ / 4), c4 = e % (C_ / 4);
            float4 k4 = ((const float4*)Kc)[e];
            ((float4*)(sK[0] + row * KSTRIDE))[c4] = k4;
            float4 v4 = ((const float4*)Vc)[e];
            ((uint32_t*)sV)[2 * e]     = (uint32_t)f2bf(v4.x) | ((uint32_t)f2bf(v4.y) << 16);
            ((uint32_t*)sV)[2 * e + 1] = (uint32_t)f2bf(v4.z) | ((uint32_t)f2bf(v4.w) << 16);
        }
#endif
        __syncthreads();

        // scores s[i][j] = sum_c wa[c]*tanh(qb[i,c]+kx[j,c]); 512 pairs / 256 thr
        for (int p = tid; p < 16 * JC; p += 256) {
            int i = p >> 5, j = p & (JC - 1);   // whole wave shares i -> sQ broadcast
            const float* q = &sQ[i * C_];
            const float* k = &sK[cur][j * KSTRIDE];
            float s = 0.0f;
#pragma unroll 4
            for (int c = 0; c < C_; c += 4) {
                float4 qv = *(const float4*)(q + c);
                float4 kv = *(const float4*)(k + c);
                float4 wv = *(const float4*)(sWa + c);
                s += wv.x * fast_tanh(qv.x + kv.x);
                s += wv.y * fast_tanh(qv.y + kv.y);
                s += wv.z * fast_tanh(qv.z + kv.z);
                s += wv.w * fast_tanh(qv.w + kv.w);
            }
            sS[i * JC + j] = s;   // +bxa omitted: constant shift cancels in softmax
        }
        __syncthreads();

        // online softmax stats per row (flash-style rescale)
        if (tid < 16) {
            int i = tid;
            float mo = sM[i], cm = mo;
            for (int j = 0; j < JC; ++j) cm = fmaxf(cm, sS[i * JC + j]);
            float scale = fast_exp(mo - cm);
            float ls = sLs[i] * scale;
            for (int j = 0; j < JC; ++j) {
                float e = fast_exp(sS[i * JC + j] - cm);
                ls += e;
                sP[i * JC + j] = f2bf(e);
            }
            sM[i] = cm; sLs[i] = ls; sScale[i] = scale;
        }
        __syncthreads();

        // acc = acc*scale + P(16x32) @ V(32x16-tile)
#pragma unroll
        for (int r = 0; r < 8; ++r) acc[r] *= sScale[r + 8 * half];

        uint32_t af[8];
        const uint32_t* sP32 = (const uint32_t*)sP;
#pragma unroll
        for (int r = 0; r < 8; ++r) {
            int k = (r < 4 ? half * 8 + 2 * r : 16 + half * 8 + 2 * (r - 4));
            af[r] = sP32[(M * JC + k) >> 1];
        }
        uint32_t bfr[8];
        const int N = wave * 16 + M;
#pragma unroll
        for (int r = 0; r < 8; ++r) {
            int k = (r < 4 ? half * 8 + 2 * r : 16 + half * 8 + 2 * (r - 4));
            uint32_t lo = sV[k * D_ + N];
            uint32_t hi = sV[(k + 1) * D_ + N];
            bfr[r] = lo | (hi << 16);
        }
        acc = __builtin_amdgcn_wmma_f32_16x16x32_bf16(
            false, pack_bf(af), false, pack_bf(bfr), (short)0, acc, false, false);
        __syncthreads();   // protect sV/sS/sP for next chunk
    }

    const int N = wave * 16 + M;
    const int rowbase = b * L_ + it * 16;
    if (nsplit == 1) {
        // out = acc / (l + eps)
#pragma unroll
        for (int r = 0; r < 8; ++r) {
            int m = r + 8 * half;
            float inv = __builtin_amdgcn_rcpf(sLs[m] + EPS_);
            out[((size_t)rowbase + m) * D_ + N] = acc[r] * inv;
        }
    } else {
        // write flash partials (unnormalized acc + per-row m, l)
#pragma unroll
        for (int r = 0; r < 8; ++r) {
            int m = r + 8 * half;
            pacc[((size_t)sp * NROWS_ + rowbase + m) * D_ + N] = acc[r];
        }
        if (tid < 16) {
            pm[sp * NROWS_ + rowbase + tid] = sM[tid];
            pl[sp * NROWS_ + rowbase + tid] = sLs[tid];
        }
    }
}

// ---------- kernel 3: merge j-split flash partials ----------
__global__ __launch_bounds__(256) void combine_kernel(
    const float* __restrict__ pacc, const float* __restrict__ pm,
    const float* __restrict__ pl, float* __restrict__ out, int nsplit) {
    int gid = blockIdx.x * 256 + threadIdx.x;   // over B*L*D
    int row = gid / D_;
    float mstar = -3.0e38f;
    for (int s = 0; s < nsplit; ++s) mstar = fmaxf(mstar, pm[s * NROWS_ + row]);
    float num = 0.0f, den = 0.0f;
    for (int s = 0; s < nsplit; ++s) {
        float sc = fast_exp(pm[s * NROWS_ + row] - mstar);
        num += sc * pacc[(size_t)s * NROWS_ * D_ + gid];
        den += sc * pl[s * NROWS_ + row];
    }
    out[gid] = num * __builtin_amdgcn_rcpf(den + EPS_);
}

// ---------- launcher ----------
extern "C" void kernel_launch(void* const* d_in, const int* in_sizes, int n_in,
                              void* d_out, int out_size, void* d_ws, size_t ws_size,
                              hipStream_t stream) {
    const float* inps  = (const float*)d_in[0];
    const float* wei_t = (const float*)d_in[1];
    const float* wei_x = (const float*)d_in[2];
    const float* bxh   = (const float*)d_in[3];
    const float* wei_a = (const float*)d_in[4];
    // d_in[5] = bxa: constant score shift, cancels in softmax -> unused.
    float* out = (float*)d_out;

    // workspace layout: qb | kx | pacc[nsplit] | pm[nsplit] | pl[nsplit]
    const size_t QK   = (size_t)NROWS_ * C_;           // floats per projection
    const int    SPLT = 4;                             // j-split for occupancy
    const size_t need4 = (2 * QK + (size_t)SPLT * NROWS_ * D_ +
                          2 * (size_t)SPLT * NROWS_) * sizeof(float);
    const int nsplit = (ws_size >= need4) ? SPLT : 1;  // deterministic in ws_size

    float* qbw  = (float*)d_ws;
    float* kxw  = qbw + QK;
    float* pacc = kxw + QK;
    float* pm   = pacc + (size_t)nsplit * NROWS_ * D_;
    float* pl   = pm + (size_t)nsplit * NROWS_;

    proj_kernel<<<NROWS_ / 16, 256, 0, stream>>>(inps, wei_t, wei_x, bxh, qbw, kxw);
    attn_kernel<<<(NROWS_ / 16) * nsplit, 256, 0, stream>>>(
        inps, qbw, kxw, wei_a, out, pacc, pm, pl, nsplit);
    if (nsplit > 1)
        combine_kernel<<<(NROWS_ * D_) / 256, 256, 0, stream>>>(pacc, pm, pl, out, nsplit);
}
